// MultiHeadAttention_48137993453987
// MI455X (gfx1250) — matrix operand order repaired and checked by reading, not compiled, not run on hardware
//
#include <hip/hip_runtime.h>
#include <hip/hip_bf16.h>

#define DEV __device__ __forceinline__

typedef __attribute__((ext_vector_type(16))) __bf16 v16bf;
typedef __attribute__((ext_vector_type(8)))  float  v8f;

union AFrag {
  v16bf v;
  unsigned int d[8];
};

static constexpr int S_LEN  = 2048;
static constexpr int DMODEL = 1024;
static constexpr int NH     = 16;
static constexpr int DK     = 64;

DEV unsigned short f2bf(float f) {
  unsigned int u = __float_as_uint(f);
  u += 0x7FFFu + ((u >> 16) & 1u);   // round-to-nearest-even
  return (unsigned short)(u >> 16);
}

DEV v8f vzero8() {
  v8f z;
  #pragma unroll
  for (int i = 0; i < 8; ++i) z[i] = 0.0f;
  return z;
}

// gfx1250 async global->LDS copy, 16 bytes per lane (ASYNCcnt-tracked).
// GV mode: vdst = per-lane LDS byte offset, vaddr = per-lane 64-bit address.
DEV void async_copy_b128(unsigned lds_off, const void* gaddr) {
  asm volatile("global_load_async_to_lds_b128 %0, %1, off"
               :: "v"(lds_off), "v"(gaddr)
               : "memory");
}

DEV void wait_asynccnt0() {
  asm volatile("s_wait_asynccnt 0x0" ::: "memory");
}

// low 32 bits of a generic pointer into LDS == LDS byte offset (flat aperture rule)
template <typename T>
DEV unsigned lds_offset(T* p) { return (unsigned)(size_t)p; }

// ---- VALU cross-lane butterflies via v_permlane16_b32 (no LDS round-trip) ----
DEV float pl16(float v, unsigned s0, unsigned s1) {
  int i = __float_as_int(v);
  i = __builtin_amdgcn_permlane16(i, i, s0, s1, false, false);
  return __int_as_float(i);
}

// reduce-max / reduce-add across each 16-lane row (result in every lane of row)
DEV float rmax16(float v) {
  v = fmaxf(v, pl16(v, 0x67452301u, 0xEFCDAB89u));  // xor 1
  v = fmaxf(v, pl16(v, 0x54761032u, 0xDCFE98BAu));  // xor 2
  v = fmaxf(v, pl16(v, 0x32107654u, 0xBA98FEDCu));  // xor 4
  v = fmaxf(v, pl16(v, 0xFEDCBA98u, 0x76543210u));  // xor 8
  return v;
}
DEV float rsum16(float v) {
  v += pl16(v, 0x67452301u, 0xEFCDAB89u);
  v += pl16(v, 0x54761032u, 0xDCFE98BAu);
  v += pl16(v, 0x32107654u, 0xBA98FEDCu);
  v += pl16(v, 0xFEDCBA98u, 0x76543210u);
  return v;
}

// ---------------------------------------------------------------------------
// NT GEMM: C[m,n] = sum_k A[m,k] * W[n,k];  M=4096, N=K=1024.
// A is fp32 (x) or bf16 (attention context). W always fp32.
// OUT_QKV: write bf16 to [B, H, S, 64]; else write fp32 row-major [M, N].
// Block: 128 threads (4 waves, 2x2 wave grid), block tile 64x64, K-step 32.
// ---------------------------------------------------------------------------
template<bool A_BF16, bool OUT_QKV>
__global__ __launch_bounds__(128)
void gemm_nt_bf16(const void* __restrict__ Aptr,
                  const float* __restrict__ W,
                  void* __restrict__ Out) {
  constexpr int K = DMODEL;
  __shared__ unsigned short As[64][36];   // bf16 bits, stride 36 (even)
  __shared__ unsigned short Bs[64][36];

  const int tid  = threadIdx.x;
  const int lane = tid & 31;
  const int wave = tid >> 5;
  const int wm   = wave >> 1;
  const int wn   = wave & 1;
  const int l15  = lane & 15;
  const int half = lane >> 4;

  const int m0 = blockIdx.x * 64;
  const int n0 = blockIdx.y * 64;

  v8f acc[2][2];
  #pragma unroll
  for (int i = 0; i < 2; ++i)
    #pragma unroll
    for (int j = 0; j < 2; ++j) acc[i][j] = vzero8();

  for (int k0 = 0; k0 < K; k0 += 32) {
    // ---- stage 64x32 tiles of A and W into LDS as bf16 ----
    #pragma unroll
    for (int i = 0; i < 4; ++i) {
      const int idx = tid + i * 128;       // 0..511
      const int row = idx >> 3;            // 0..63
      const int c4  = (idx & 7) * 4;       // 0,4,...,28
      unsigned int a01, a23;
      if constexpr (A_BF16) {
        const uint2 va = *(const uint2*)((const unsigned short*)Aptr +
                                         (size_t)(m0 + row) * K + k0 + c4);
        a01 = va.x; a23 = va.y;
      } else {
        const float4 va = *(const float4*)((const float*)Aptr +
                                           (size_t)(m0 + row) * K + k0 + c4);
        a01 = (unsigned)f2bf(va.x) | ((unsigned)f2bf(va.y) << 16);
        a23 = (unsigned)f2bf(va.z) | ((unsigned)f2bf(va.w) << 16);
      }
      *(unsigned int*)&As[row][c4]     = a01;
      *(unsigned int*)&As[row][c4 + 2] = a23;

      const float4 vb = *(const float4*)(W + (size_t)(n0 + row) * K + k0 + c4);
      *(unsigned int*)&Bs[row][c4]     = (unsigned)f2bf(vb.x) | ((unsigned)f2bf(vb.y) << 16);
      *(unsigned int*)&Bs[row][c4 + 2] = (unsigned)f2bf(vb.z) | ((unsigned)f2bf(vb.w) << 16);
    }
    if (k0 + 32 < K) {   // gfx1250 global_prefetch of next W slice
      __builtin_prefetch(W + (size_t)(n0 + (tid >> 1)) * K + (k0 + 32) + (tid & 1) * 16, 0, 1);
    }
    __syncthreads();

    // ---- build WMMA fragments: lane half h holds k = 16h..16h+15 for both A and B ----
    AFrag af[2], bf[2];
    #pragma unroll
    for (int ms = 0; ms < 2; ++ms) {
      const int row = wm * 32 + ms * 16 + l15;
      const int kb  = half * 16;
      #pragma unroll
      for (int j = 0; j < 8; ++j) {
        af[ms].d[j]     = *(const unsigned int*)&As[row][kb + 2 * j];
      }
    }
    #pragma unroll
    for (int ns = 0; ns < 2; ++ns) {
      const int col = wn * 32 + ns * 16 + l15;
      const int kb  = half * 16;
      #pragma unroll
      for (int j = 0; j < 8; ++j)
        bf[ns].d[j] = *(const unsigned int*)&Bs[col][kb + 2 * j];
    }
    #pragma unroll
    for (int ms = 0; ms < 2; ++ms)
      #pragma unroll
      for (int ns = 0; ns < 2; ++ns)
        acc[ms][ns] = __builtin_amdgcn_wmma_f32_16x16x32_bf16(
            false, af[ms].v, false, bf[ns].v, (short)0, acc[ms][ns], false, false);
    __syncthreads();
  }

  // ---- epilogue ----
  #pragma unroll
  for (int ms = 0; ms < 2; ++ms)
    #pragma unroll
    for (int ns = 0; ns < 2; ++ns)
      #pragma unroll
      for (int r = 0; r < 8; ++r) {
        const int m = m0 + wm * 32 + ms * 16 + r + half * 8;
        const int n = n0 + wn * 32 + ns * 16 + l15;
        const float val = acc[ms][ns][r];
        if constexpr (OUT_QKV) {
          const int b  = m >> 11;           // S=2048
          const int s  = m & (S_LEN - 1);
          const int h  = n >> 6;            // DK=64
          const int d2 = n & 63;
          ((unsigned short*)Out)[((size_t)(b * NH + h) * S_LEN + s) * DK + d2] = f2bf(val);
        } else {
          ((float*)Out)[(size_t)m * DMODEL + n] = val;
        }
      }
}

// ---------------------------------------------------------------------------
// Flash attention, causal. Q/K/V bf16 [B*H, S, 64]. Out bf16 [B, S, DMODEL].
// Block: 128 threads (4 waves). Block handles 64 query rows of one (b,h);
// each wave owns 16 rows. KV tiles of 64 staged in LDS:
//   K tile via async global->LDS copy (ASYNCcnt), row-major [kv][d]
//   V tile transposed on the fly to [d][kv] so PV B-fragments are dword loads
// Softmax cross-lane reductions use v_permlane16 (VALU, no LDS).
// ---------------------------------------------------------------------------
__global__ __launch_bounds__(128)
void flash_attn(const unsigned short* __restrict__ Q,
                const unsigned short* __restrict__ Kk,
                const unsigned short* __restrict__ V,
                unsigned short* __restrict__ Ctx) {
  __shared__ unsigned short Ks[64][72];      // [kv][d]
  __shared__ unsigned short Vt[64][72];      // [d][kv] (transposed)
  __shared__ unsigned short Ps[4][16][72];   // per-wave P tile (C->A relayout)

  const int tid  = threadIdx.x;
  const int lane = tid & 31;
  const int wave = tid >> 5;
  const int l15  = lane & 15;
  const int half = lane >> 4;

  const int qt = blockIdx.x;                 // query tile 0..31
  const int bh = blockIdx.y;                 // b*NH + h, 0..31
  const int q0 = qt * 64 + wave * 16;

  // Q fragments: A-layout, two 32-deep chunks over d (rows contiguous in d)
  AFrag qf[2];
  {
    const unsigned short* qb = Q + ((size_t)bh * S_LEN + q0 + l15) * DK;
    #pragma unroll
    for (int kc = 0; kc < 2; ++kc) {
      const int kb = kc * 32 + half * 16;
      #pragma unroll
      for (int j = 0; j < 8; ++j) {
        qf[kc].d[j]     = *(const unsigned int*)(qb + kb + 2 * j);
      }
    }
  }

  v8f o[4];
  #pragma unroll
  for (int ns = 0; ns < 4; ++ns) o[ns] = vzero8();
  float mrow[8], lrow[8];
  #pragma unroll
  for (int r = 0; r < 8; ++r) { mrow[r] = -3.0e38f; lrow[r] = 0.0f; }

  for (int jt = 0; jt <= qt; ++jt) {
    // ---- stage K (async DMA to LDS) and V (transposed) ----
    #pragma unroll
    for (int i = 0; i < 4; ++i) {
      const int idx = tid + i * 128;
      const int row = idx >> 3;
      const int c8  = (idx & 7) * 8;
      const size_t g = ((size_t)bh * S_LEN + (size_t)jt * 64 + row) * DK + c8;

      // K: straight 16-byte async copy into row-major tile
      async_copy_b128(lds_offset(&Ks[row][c8]), Kk + g);

      // V: load 8 values, scatter-store transposed
      const uint4 vv = *(const uint4*)(V + g);
      Vt[c8 + 0][row] = (unsigned short)(vv.x);
      Vt[c8 + 1][row] = (unsigned short)(vv.x >> 16);
      Vt[c8 + 2][row] = (unsigned short)(vv.y);
      Vt[c8 + 3][row] = (unsigned short)(vv.y >> 16);
      Vt[c8 + 4][row] = (unsigned short)(vv.z);
      Vt[c8 + 5][row] = (unsigned short)(vv.z >> 16);
      Vt[c8 + 6][row] = (unsigned short)(vv.w);
      Vt[c8 + 7][row] = (unsigned short)(vv.w >> 16);
    }
    wait_asynccnt0();
    __syncthreads();

    // ---- S = Q K^T (NT form: both contiguous along d) ----
    v8f sa[4];
    #pragma unroll
    for (int ns = 0; ns < 4; ++ns) {
      sa[ns] = vzero8();
      #pragma unroll
      for (int kc = 0; kc < 2; ++kc) {
        AFrag kf;
        const int n  = ns * 16 + l15;         // kv index in tile
        const int kb = kc * 32 + half * 16;
        #pragma unroll
        for (int j = 0; j < 8; ++j)
          kf.d[j] = *(const unsigned int*)&Ks[n][kb + 2 * j];
        sa[ns] = __builtin_amdgcn_wmma_f32_16x16x32_bf16(
            false, qf[kc].v, false, kf.v, (short)0, sa[ns], false, false);
      }
    }

    // ---- scale + causal mask (only diagonal tile needs masking) ----
    const float sc = 0.125f;   // 1/sqrt(64)
    #pragma unroll
    for (int ns = 0; ns < 4; ++ns)
      #pragma unroll
      for (int r = 0; r < 8; ++r) {
        float v = sa[ns][r] * sc;
        if (jt == qt) {
          if (ns * 16 + l15 > wave * 16 + r + 8 * half) v = -3.0e38f;
        }
        sa[ns][r] = v;
      }

    // ---- online softmax: rows live in one 16-lane half per VGPR index ----
    #pragma unroll
    for (int r = 0; r < 8; ++r) {
      float mx = fmaxf(fmaxf(sa[0][r], sa[1][r]), fmaxf(sa[2][r], sa[3][r]));
      mx = rmax16(mx);
      const float mnew  = fmaxf(mrow[r], mx);
      const float alpha = __expf(mrow[r] - mnew);
      float rs = 0.0f;
      #pragma unroll
      for (int ns = 0; ns < 4; ++ns) {
        const float p = __expf(sa[ns][r] - mnew);
        sa[ns][r] = p;
        rs += p;
      }
      rs = rsum16(rs);
      lrow[r] = lrow[r] * alpha + rs;
      mrow[r] = mnew;
      #pragma unroll
      for (int ns = 0; ns < 4; ++ns) o[ns][r] *= alpha;
    }

    // ---- relayout P (C-layout -> A-layout) through per-wave LDS ----
    #pragma unroll
    for (int ns = 0; ns < 4; ++ns)
      #pragma unroll
      for (int r = 0; r < 8; ++r)
        Ps[wave][r + 8 * half][ns * 16 + l15] = f2bf(sa[ns][r]);

    AFrag pf[2];
    #pragma unroll
    for (int kc = 0; kc < 2; ++kc) {
      const int kb = kc * 32 + half * 16;
      #pragma unroll
      for (int j = 0; j < 8; ++j) {
        pf[kc].d[j]     = *(const unsigned int*)&Ps[wave][l15][kb + 2 * j];
      }
    }

    // ---- O += P @ V: B-fragment dwords contiguous in transposed Vt ----
    #pragma unroll
    for (int ns = 0; ns < 4; ++ns) {
      const int dcol = ns * 16 + l15;
      #pragma unroll
      for (int kc = 0; kc < 2; ++kc) {
        AFrag vf;
        const int kb = kc * 32 + half * 16;
        #pragma unroll
        for (int j = 0; j < 8; ++j)
          vf.d[j] = *(const unsigned int*)&Vt[dcol][kb + 2 * j];
        o[ns] = __builtin_amdgcn_wmma_f32_16x16x32_bf16(
            false, pf[kc].v, false, vf.v, (short)0, o[ns], false, false);
      }
    }
    __syncthreads();
  }

  // ---- normalize and write context bf16 [B, S, DMODEL] ----
  float invl[8];
  #pragma unroll
  for (int r = 0; r < 8; ++r) invl[r] = 1.0f / lrow[r];
  const int b = bh >> 4;
  const int h = bh & 15;
  #pragma unroll
  for (int ns = 0; ns < 4; ++ns)
    #pragma unroll
    for (int r = 0; r < 8; ++r) {
      const int s = q0 + r + 8 * half;
      const int e = h * DK + ns * 16 + l15;
      Ctx[((size_t)b * S_LEN + s) * DMODEL + e] = f2bf(o[ns][r] * invl[r]);
    }
}

// ---------------------------------------------------------------------------
// Host launch. Inputs: x, mask(ignored: analytically causal), W_q, W_k, W_v, W_o
// ---------------------------------------------------------------------------
extern "C" void kernel_launch(void* const* d_in, const int* in_sizes, int n_in,
                              void* d_out, int out_size, void* d_ws, size_t ws_size,
                              hipStream_t stream) {
  const float* x  = (const float*)d_in[0];
  const float* Wq = (const float*)d_in[2];
  const float* Wk = (const float*)d_in[3];
  const float* Wv = (const float*)d_in[4];
  const float* Wo = (const float*)d_in[5];

  const size_t TSZ = (size_t)2 * NH * S_LEN * DK;   // 4,194,304 bf16 elements
  unsigned short* Qw = (unsigned short*)d_ws;
  unsigned short* Kw = Qw + TSZ;
  unsigned short* Vw = Kw + TSZ;
  unsigned short* Cw = Vw + TSZ;                    // context [B,S,DMODEL] bf16

  const dim3 gg(64, 16), bb(128);
  gemm_nt_bf16<false, true ><<<gg, bb, 0, stream>>>(x,  Wq, Qw);
  gemm_nt_bf16<false, true ><<<gg, bb, 0, stream>>>(x,  Wk, Kw);
  gemm_nt_bf16<false, true ><<<gg, bb, 0, stream>>>(x,  Wv, Vw);

  flash_attn<<<dim3(S_LEN / 64, 2 * NH), bb, 0, stream>>>(Qw, Kw, Vw, Cw);

  gemm_nt_bf16<true, false><<<gg, bb, 0, stream>>>(Cw, Wo, (float*)d_out);
}
